// GAT_loop_70970039599312
// MI455X (gfx1250) — matrix operand order repaired and checked by reading, not compiled, not run on hardware
//
#include <hip/hip_runtime.h>

typedef __attribute__((ext_vector_type(16))) _Float16 v16h;
typedef __attribute__((ext_vector_type(8)))  _Float16 v8h;
typedef __attribute__((ext_vector_type(8)))  float    v8f;

#define HN 4      // heads
#define CC 64     // channels per head
#define HC 256    // HN*CC

__device__ inline float leakyf(float x, float s) { return x > 0.f ? x : s * x; }

__device__ inline void atomicMaxF(float* addr, float val) {
  unsigned int* ua = (unsigned int*)addr;
  unsigned int old = *ua;
  while (__uint_as_float(old) < val) {
    unsigned int assumed = old;
    old = atomicCAS(ua, assumed, __float_as_uint(val));
    if (old == assumed) break;
  }
}

// ---------------------------------------------------------------------------
// f32 -> f16 copy (coalesced)
// ---------------------------------------------------------------------------
__global__ void cvt_f16(const float* __restrict__ in, _Float16* __restrict__ out, long n) {
  long i = (long)blockIdx.x * blockDim.x + threadIdx.x;
  if (i < n) out[i] = (_Float16)in[i];
}

// B[K,Nn] (f32, row-major) -> BT[Nn,K] (f16, row-major); tiny matrices
__global__ void transpose_cvt_f16(const float* __restrict__ B, _Float16* __restrict__ BT,
                                  int K, int Nn) {
  int idx = blockIdx.x * blockDim.x + threadIdx.x;
  if (idx >= K * Nn) return;
  int n = idx / K, k = idx - n * K;        // writes coalesced along K
  BT[idx] = (_Float16)B[(long)k * Nn + n];
}

// ---------------------------------------------------------------------------
// WMMA GEMM: D[M,Nn] = A[M,K] * B[K,Nn]; A is f16 row-major, BT is f16 B^T
// [Nn,K] row-major, D is f32. One wave computes a 16x64 strip (4 accums,
// A-fragment reused 4x). Requires M%16==0, K%32==0, Nn%64==0.
// All fragment loads are contiguous 16B/32B vector loads (global_load_b128).
// ---------------------------------------------------------------------------
__global__ void wmma_gemm(const _Float16* __restrict__ A, const _Float16* __restrict__ BT,
                          float* __restrict__ D, int M, int K, int Nn, int waves_total) {
  int gid  = blockIdx.x * blockDim.x + threadIdx.x;
  int wave = gid >> 5;
  int lane = gid & 31;
  if (wave >= waves_total) return;          // uniform per wave: EXEC stays all-ones
  int ntn   = Nn >> 6;
  int tm    = wave / ntn;
  int tn    = wave - tm * ntn;
  int l16   = lane & 15;
  int khalf = lane >> 4;                    // which K-half this lane holds

  // A fragment (16x32 f16): halves i -> K = 16*khalf + i
  const _Float16* arow = A + (long)(tm * 16 + l16) * K + khalf * 16;
  // B fragment (32x16 f16): halves i -> K = 16*khalf + i, for column tn*64+t*16+l16
  const _Float16* bp   = BT + (long)(tn * 64 + l16) * K + khalf * 16;
  const long bstride   = (long)16 * K;      // advance one 16-col tile in BT

  v8f acc[4] = {v8f{}, v8f{}, v8f{}, v8f{}};

  for (int k0 = 0; k0 < K; k0 += 32) {
    v8h alo = *(const v8h*)(arow + k0);
    v8h ahi = *(const v8h*)(arow + k0 + 8);
    v16h a  = __builtin_shufflevector(alo, ahi,
                                      0, 1, 2, 3, 4, 5, 6, 7,
                                      8, 9, 10, 11, 12, 13, 14, 15);
#pragma unroll
    for (int t = 0; t < 4; ++t) {
      v16h b = *(const v16h*)(bp + t * bstride + k0);
      acc[t] = __builtin_amdgcn_wmma_f32_16x16x32_f16(false, a, false, b,
                                                      (short)0, acc[t], false, false);
    }
  }

  // C/D layout: VGPR r, lanes 0-15 -> M=r, lanes 16-31 -> M=8+r; N = lane&15
  int rowbase = tm * 16 + khalf * 8;
#pragma unroll
  for (int t = 0; t < 4; ++t) {
    int col = tn * 64 + t * 16 + l16;
#pragma unroll
    for (int r = 0; r < 8; ++r)
      D[(long)(rowbase + r) * Nn + col] = acc[t][r];
  }
}

// ---------------------------------------------------------------------------
__global__ void fill_f32(float* __restrict__ p, float v, long n) {
  long i = (long)blockIdx.x * blockDim.x + threadIdx.x;
  if (i < n) p[i] = v;
}

// al_s[n,h] = sum_c h[n,h,c]*a_src[h,c] ; same for al_d
__global__ void attn_scores(const float* __restrict__ h, const float* __restrict__ a_src,
                            const float* __restrict__ a_dst, float* __restrict__ al_s,
                            float* __restrict__ al_d, int N) {
  int idx = blockIdx.x * blockDim.x + threadIdx.x;
  if (idx >= N * HN) return;
  int hd = idx & (HN - 1);
  int n  = idx >> 2;
  const float* hp = h + (long)n * HC + hd * CC;
  const float* as = a_src + hd * CC;
  const float* ad = a_dst + hd * CC;
  float ss = 0.f, sd = 0.f;
#pragma unroll 8
  for (int c = 0; c < CC; ++c) { float v = hp[c]; ss += v * as[c]; sd += v * ad[c]; }
  al_s[idx] = ss;
  al_d[idx] = sd;
}

// pass 1: alpha = leaky(al_s[src]+al_d[dst], 0.2); segment max over dst
__global__ void edge_max(const int* __restrict__ srcI, const int* __restrict__ dstI,
                         int E, int Et, const float* __restrict__ al_s,
                         const float* __restrict__ al_d, float* __restrict__ alpha,
                         float* __restrict__ mseg) {
  int idx = blockIdx.x * blockDim.x + threadIdx.x;
  if (idx >= Et * HN) return;
  int e = idx >> 2, hd = idx & 3;
  int s, d;
  if (e < E) { s = srcI[e]; d = dstI[e]; } else { s = d = e - E; }  // self-loops
  float a = leakyf(al_s[s * HN + hd] + al_d[d * HN + hd], 0.2f);
  alpha[idx] = a;
  atomicMaxF(&mseg[d * HN + hd], a);
}

// pass 2: e = exp(alpha - m[dst]); segment sum over dst; alpha <- e
__global__ void edge_expsum(const int* __restrict__ dstI, int E, int Et,
                            const float* __restrict__ mseg, float* __restrict__ alpha,
                            float* __restrict__ sseg) {
  int idx = blockIdx.x * blockDim.x + threadIdx.x;
  if (idx >= Et * HN) return;
  int e = idx >> 2, hd = idx & 3;
  int d = (e < E) ? dstI[e] : e - E;
  float ex = __expf(alpha[idx] - mseg[d * HN + hd]);
  alpha[idx] = ex;
  atomicAdd(&sseg[d * HN + hd], ex);
}

// pass 3: accum[dst] += h[src] * (e / (s[dst]+eps)); one thread per (edge,h,c)
__global__ void edge_message(const int* __restrict__ srcI, const int* __restrict__ dstI,
                             int E, long EtHC, const float* __restrict__ h,
                             const float* __restrict__ alpha, const float* __restrict__ sseg,
                             float* __restrict__ accum) {
  long idx = (long)blockIdx.x * blockDim.x + threadIdx.x;
  if (idx >= EtHC) return;
  int e   = (int)(idx >> 8);
  int rem = (int)(idx & 255);
  int hd  = rem >> 6;
  int s, d;
  if (e < E) { s = srcI[e]; d = dstI[e]; } else { s = d = e - E; }
  float coef = alpha[e * HN + hd] / (sseg[d * HN + hd] + 1e-16f);
  atomicAdd(&accum[(long)d * HC + rem], h[(long)s * HC + rem] * coef);
}

// mean over heads + bias + leaky(0.01)
__global__ void node_finalize(const float* __restrict__ accum, const float* __restrict__ bias,
                              float* __restrict__ xout, int N) {
  int idx = blockIdx.x * blockDim.x + threadIdx.x;
  if (idx >= N * CC) return;
  int n = idx >> 6, c = idx & 63;
  const float* ap = accum + (long)n * HC + c;
  float v = (ap[0] + ap[64] + ap[128] + ap[192]) * 0.25f + bias[c];
  xout[idx] = leakyf(v, 0.01f);
}

__global__ void pool_scatter(const float* __restrict__ x, const int* __restrict__ batch,
                             float* __restrict__ gmp, float* __restrict__ gsum, int N) {
  int idx = blockIdx.x * blockDim.x + threadIdx.x;
  if (idx >= N * CC) return;
  int n = idx >> 6, c = idx & 63;
  int g = batch[n];
  float v = x[idx];
  atomicMaxF(&gmp[g * CC + c], v);
  atomicAdd(&gsum[g * CC + c], v);
}

__global__ void pool_count(const int* __restrict__ batch, float* __restrict__ cnt, int N) {
  int n = blockIdx.x * blockDim.x + threadIdx.x;
  if (n < N) atomicAdd(&cnt[batch[n]], 1.f);
}

__global__ void pool_hid(const float* __restrict__ gmp, const float* __restrict__ gsum,
                         const float* __restrict__ cnt, float* __restrict__ hid, int G) {
  int idx = blockIdx.x * blockDim.x + threadIdx.x;
  if (idx >= G * 2 * CC) return;
  int g = idx >> 7, j = idx & 127;
  hid[idx] = (j < CC) ? gmp[g * CC + j] : gsum[g * CC + (j - CC)] / fmaxf(cnt[g], 1.f);
}

__global__ void bias_leaky(float* __restrict__ p, const float* __restrict__ b, int total, int cols) {
  int idx = blockIdx.x * blockDim.x + threadIdx.x;
  if (idx >= total) return;
  p[idx] = leakyf(p[idx] + b[idx % cols], 0.01f);
}

__global__ void final_dot(const float* __restrict__ r1, const float* __restrict__ w2,
                          const float* __restrict__ b2, float* __restrict__ out, int G) {
  int g = blockIdx.x * blockDim.x + threadIdx.x;
  if (g >= G) return;
  float s = 0.f;
#pragma unroll 8
  for (int c = 0; c < CC; ++c) s += r1[g * CC + c] * w2[c];
  out[g] = s + b2[0];
}

// ---------------------------------------------------------------------------
static inline int cdiv(long a, long b) { return (int)((a + b - 1) / b); }

extern "C" void kernel_launch(void* const* d_in, const int* in_sizes, int n_in,
                              void* d_out, int out_size, void* d_ws, size_t ws_size,
                              hipStream_t stream) {
  const float* x     = (const float*)d_in[0];
  const int*   ei    = (const int*)d_in[1];
  const int*   batch = (const int*)d_in[3];
  const float* W0    = (const float*)d_in[4];
  const float* as0   = (const float*)d_in[5];
  const float* ad0   = (const float*)d_in[6];
  const float* b0    = (const float*)d_in[7];
  const float* W1    = (const float*)d_in[8];
  const float* as1   = (const float*)d_in[9];
  const float* ad1   = (const float*)d_in[10];
  const float* b1    = (const float*)d_in[11];
  const float* W2    = (const float*)d_in[12];
  const float* as2   = (const float*)d_in[13];
  const float* ad2   = (const float*)d_in[14];
  const float* b2    = (const float*)d_in[15];
  const float* Wr1   = (const float*)d_in[16];
  const float* br1   = (const float*)d_in[17];
  const float* Wr2   = (const float*)d_in[18];
  const float* br2   = (const float*)d_in[19];

  const int N  = in_sizes[0] / 128;   // 50000
  const int E  = in_sizes[1] / 2;     // 800000
  const int Et = E + N;               // edges + self-loops
  const int G  = out_size;            // 512

  const int* srcI = ei;               // edge_index[0,:]
  const int* dstI = ei + E;           // edge_index[1,:]

  // workspace carve-out; every block below is a multiple of 32 bytes,
  // so all f16 vector loads stay 32B-aligned.
  float* ws    = (float*)d_ws;
  float* h     = ws; ws += (long)N * HC;        // [N,256] f32
  float* accum = ws; ws += (long)N * HC;        // [N,256] f32
  float* xbuf  = ws; ws += (long)N * CC;        // [N,64]  f32
  float* al_s  = ws; ws += (long)N * HN;
  float* al_d  = ws; ws += (long)N * HN;
  float* mseg  = ws; ws += (long)N * HN;
  float* sseg  = ws; ws += (long)N * HN;
  float* alpha = ws; ws += (long)Et * HN;
  float* gmp   = ws; ws += (long)G * CC;
  float* gsum  = ws; ws += (long)G * CC;
  float* cnt   = ws; ws += G;
  float* hid   = ws; ws += (long)G * 2 * CC;
  float* r1    = ws; ws += (long)G * CC;
  _Float16* a16  = (_Float16*)ws;               // max N*128 halfs (12.8 MB)
  ws += (long)N * 64;                           // N*128 halfs == N*64 floats
  _Float16* bt16 = (_Float16*)ws;               // max 256*128 halfs
  ws += (long)(HC * 128) / 2;

  const int T = 256;
  const float NEG_INF = -__builtin_inff();

  const float* Wl[3]  = {W0, W1, W2};
  const float* asl[3] = {as0, as1, as2};
  const float* adl[3] = {ad0, ad1, ad2};
  const float* bl[3]  = {b0, b1, b2};

  const float* xin = x;
  int Kin = 128;
  for (int L = 0; L < 3; ++L) {
    // --- h = xin @ W  via WMMA (f16 operands, f32 accumulate) ---
    cvt_f16<<<cdiv((long)N * Kin, T), T, 0, stream>>>(xin, a16, (long)N * Kin);
    transpose_cvt_f16<<<cdiv((long)Kin * HC, T), T, 0, stream>>>(Wl[L], bt16, Kin, HC);
    int waves = (N / 16) * (HC / 64);
    wmma_gemm<<<cdiv((long)waves * 32, T), T, 0, stream>>>(a16, bt16, h, N, Kin, HC, waves);

    // --- attention + segment softmax + message aggregation ---
    attn_scores<<<cdiv((long)N * HN, T), T, 0, stream>>>(h, asl[L], adl[L], al_s, al_d, N);
    fill_f32<<<cdiv((long)N * HN, T), T, 0, stream>>>(mseg, NEG_INF, (long)N * HN);
    fill_f32<<<cdiv((long)N * HN, T), T, 0, stream>>>(sseg, 0.f, (long)N * HN);
    fill_f32<<<cdiv((long)N * HC, T), T, 0, stream>>>(accum, 0.f, (long)N * HC);
    edge_max<<<cdiv((long)Et * HN, T), T, 0, stream>>>(srcI, dstI, E, Et, al_s, al_d, alpha, mseg);
    edge_expsum<<<cdiv((long)Et * HN, T), T, 0, stream>>>(dstI, E, Et, mseg, alpha, sseg);
    edge_message<<<cdiv((long)Et * HC, T), T, 0, stream>>>(srcI, dstI, E, (long)Et * HC,
                                                           h, alpha, sseg, accum);
    node_finalize<<<cdiv((long)N * CC, T), T, 0, stream>>>(accum, bl[L], xbuf, N);
    xin = xbuf;
    Kin = CC;
  }

  // --- graph pooling (max + mean) ---
  fill_f32<<<cdiv((long)G * CC, T), T, 0, stream>>>(gmp, NEG_INF, (long)G * CC);
  fill_f32<<<cdiv((long)G * CC, T), T, 0, stream>>>(gsum, 0.f, (long)G * CC);
  fill_f32<<<cdiv((long)G, T), T, 0, stream>>>(cnt, 0.f, (long)G);
  pool_scatter<<<cdiv((long)N * CC, T), T, 0, stream>>>(xbuf, batch, gmp, gsum, N);
  pool_count<<<cdiv((long)N, T), T, 0, stream>>>(batch, cnt, N);
  pool_hid<<<cdiv((long)G * 2 * CC, T), T, 0, stream>>>(gmp, gsum, cnt, hid, G);

  // --- readout MLP: r1 = leaky(hid @ Wr1 + br1); out = r1 @ Wr2 + br2 ---
  cvt_f16<<<cdiv((long)G * 2 * CC, T), T, 0, stream>>>(hid, a16, (long)G * 2 * CC);
  transpose_cvt_f16<<<cdiv((long)2 * CC * CC, T), T, 0, stream>>>(Wr1, bt16, 2 * CC, CC);
  {
    int waves = (G / 16) * (CC / 64);
    wmma_gemm<<<cdiv((long)waves * 32, T), T, 0, stream>>>(a16, bt16, r1, G, 2 * CC, CC, waves);
  }
  bias_leaky<<<cdiv((long)G * CC, T), T, 0, stream>>>(r1, br1, G * CC, CC);
  final_dot<<<cdiv((long)G, T), T, 0, stream>>>(r1, Wr2, br2, (float*)d_out, G);
}
